// Qwen35FullAttention_67207648248340
// MI455X (gfx1250) — matrix operand order repaired and checked by reading, not compiled, not run on hardware
//
#include <hip/hip_runtime.h>
#include <hip/hip_bf16.h>

typedef __attribute__((ext_vector_type(16))) __bf16 v16bf;
typedef __attribute__((ext_vector_type(8)))  __bf16 v8bf;
typedef __attribute__((ext_vector_type(8)))  float  v8f;
typedef __attribute__((ext_vector_type(4)))  unsigned u32x4;
typedef __attribute__((ext_vector_type(8)))  int      i32x8;
typedef __attribute__((ext_vector_type(4)))  int      i32x4;

union V16 { v16bf v; v8bf h[2]; };

#define WMMA_BF16(A,B,C) \
  __builtin_amdgcn_wmma_f32_16x16x32_bf16(false,(A),false,(B),(short)0,(C),false,false)

#if defined(__has_builtin)
#if __has_builtin(__builtin_amdgcn_tensor_load_to_lds) && \
    __has_builtin(__builtin_amdgcn_s_wait_tensorcnt)
#define HAVE_TDM 1
#endif
#endif

// ---------------------------------------------------------------- fp32 -> bf16
__global__ void cvt_f32_bf16(const float* __restrict__ in, __bf16* __restrict__ out, long n) {
  long i = (long)blockIdx.x * blockDim.x + threadIdx.x;
  long stride = (long)gridDim.x * blockDim.x;
  for (; i < n; i += stride) out[i] = (__bf16)in[i];
}

// ------------------------------------------------- generic NT GEMM (bf16 WMMA)
// C[M,N] = A[M,K] * B[N,K]^T ; A,B row-major bf16 (K contiguous on both).
// 256 threads = 8 waves (2x4). Wave tile 32x32 (4 x 16x16 WMMA). Block 64x128.
__global__ __launch_bounds__(256) void gemm_nt_bf16(
    const __bf16* __restrict__ A, const __bf16* __restrict__ B,
    float* __restrict__ Cf, __bf16* __restrict__ Cb, int M, int N, int K)
{
  const int lane = threadIdx.x & 31;
  const int wave = threadIdx.x >> 5;
  const int wm = wave >> 2, wn = wave & 3;
  const int m0 = blockIdx.y * 64 + wm * 32;
  const int n0 = blockIdx.x * 128 + wn * 32;
  const int r16 = lane & 15;
  const int hi  = lane >> 4;
  const int ka  = hi * 16;   // A: 16 contiguous halves per lane-half
  const int kb  = hi * 16;   // B: 16 contiguous halves per lane-half

  v8f acc[2][2] = {};
  for (int k = 0; k < K; k += 32) {
    V16 a[2], b[2];
#pragma unroll
    for (int i = 0; i < 2; ++i) {
      const __bf16* pa = A + (size_t)(m0 + i * 16 + r16) * K + k + ka;
      a[i].h[0] = *(const v8bf*)pa;
      a[i].h[1] = *(const v8bf*)(pa + 8);
      const __bf16* pb = B + (size_t)(n0 + i * 16 + r16) * K + k + kb;
      b[i].h[0] = *(const v8bf*)pb;
      b[i].h[1] = *(const v8bf*)(pb + 8);
      __builtin_prefetch(pa + 64, 0, 1);
      __builtin_prefetch(pb + 64, 0, 1);
    }
#pragma unroll
    for (int i = 0; i < 2; ++i)
#pragma unroll
      for (int j = 0; j < 2; ++j)
        acc[i][j] = WMMA_BF16(a[i].v, b[j].v, acc[i][j]);
  }
#pragma unroll
  for (int i = 0; i < 2; ++i)
#pragma unroll
    for (int j = 0; j < 2; ++j) {
      const int cn = n0 + j * 16 + r16;
      const int rb = m0 + i * 16 + hi * 8;
#pragma unroll
      for (int v = 0; v < 8; ++v) {
        size_t idx = (size_t)(rb + v) * N + cn;
        if (Cf) Cf[idx] = acc[i][j][v];
        else    Cb[idx] = (__bf16)acc[i][j][v];
      }
    }
}

// ------------------------------------- per-head RMSNorm + RoPE + relayout
__global__ __launch_bounds__(256) void normrope_kernel(
    const __bf16* __restrict__ src, const float* __restrict__ normw,
    const float* __restrict__ rope, __bf16* __restrict__ dst,
    int nheads, int rowStride, int headStride, int mode, int S)
{
  const int gw   = blockIdx.x * (blockDim.x >> 5) + (threadIdx.x >> 5);
  const int lane = threadIdx.x & 31;
  const int t = gw / nheads;
  const int h = gw % nheads;
  const int b = t / S, s = t % S;
  const __bf16* p = src + (size_t)t * rowStride + h * headStride + lane * 8;
  float x[8];
#pragma unroll
  for (int j = 0; j < 8; ++j) x[j] = (float)p[j];
  if (mode) {
    float ss = 0.f;
#pragma unroll
    for (int j = 0; j < 8; ++j) ss += x[j] * x[j];
#pragma unroll
    for (int m = 16; m >= 1; m >>= 1) ss += __shfl_xor(ss, m, 32);
    const float r = rsqrtf(ss * (1.0f / 256.0f) + 1e-6f);
#pragma unroll
    for (int j = 0; j < 8; ++j) x[j] = x[j] * r * (1.0f + normw[lane * 8 + j]);
    // RoPE on first 64 dims (lanes 0..7); rotate-half partner is lane^4
#pragma unroll
    for (int j = 0; j < 8; ++j) {
      float pj = __shfl_xor(x[j], 4, 32);
      if (lane < 8) {
        int d = lane * 8 + j;
        float rot = (lane < 4) ? -pj : pj;
        float c  = rope[(size_t)s * 128 + d];
        float si = rope[(size_t)s * 128 + 64 + d];
        x[j] = x[j] * c + rot * si;
      }
    }
  }
  __bf16* q = dst + ((size_t)(b * nheads + h) * S + s) * 256 + lane * 8;
#pragma unroll
  for (int j = 0; j < 8; ++j) q[j] = (__bf16)x[j];
}

// --------------------------------------------- flash attention (WMMA, wave32)
// block = 128 threads = 4 waves sharing (b,h); wave owns 16 q rows.
// K chunk staged via Tensor Data Mover (tensor_load_to_lds) when available.
__global__ __launch_bounds__(128, 1) void attn_kernel(
    const __bf16* __restrict__ qp, const __bf16* __restrict__ kp,
    const __bf16* __restrict__ vp, const __bf16* __restrict__ qgb,
    __bf16* __restrict__ yb, int S)
{
  __shared__ __bf16 ksh[32 * 256];
  __shared__ __bf16 vsh[256 * 32];
  __shared__ __bf16 psh[4][16 * 32];

  const int tid  = threadIdx.x;
  const int lane = tid & 31;
  const int wave = tid >> 5;
  const int bh = blockIdx.x;
  const int b = bh >> 4, h = bh & 15, kv = h >> 2;
  const int q0 = blockIdx.y * 64 + wave * 16;
  const int r16 = lane & 15, hi = lane >> 4;

  const __bf16* qbase = qp + (size_t)(b * 16 + h)  * S * 256;
  const __bf16* kbase = kp + (size_t)(b * 4 + kv) * S * 256;
  const __bf16* vbase = vp + (size_t)(b * 4 + kv) * S * 256;

  // preload q A-fragments: 8 x (16 rows x 32 dims)
  V16 qA[8];
#pragma unroll
  for (int kk = 0; kk < 8; ++kk) {
    const __bf16* pq = qbase + (size_t)(q0 + r16) * 256 + kk * 32 + hi * 16;
    qA[kk].h[0] = *(const v8bf*)pq;
    qA[kk].h[1] = *(const v8bf*)(pq + 8);
  }

  v8f accY[16] = {};
  float mrun[8], lrun[8];
#pragma unroll
  for (int i = 0; i < 8; ++i) { mrun[i] = -1e30f; lrun[i] = 0.f; }
  const float scale = 0.0625f;  // 256^-0.5

#ifdef HAVE_TDM
  const unsigned ksh_lds = (unsigned)(uintptr_t)(&ksh[0]);  // low 32b = LDS offset
#endif

  for (int kc = 0; kc < S; kc += 32) {
#ifdef HAVE_TDM
    // ---- stage K chunk [32 keys][256] (16 KB contiguous) via TDM ----
    if (wave == 0) {
      unsigned long long ga = (unsigned long long)(uintptr_t)(kbase + (size_t)kc * 256);
      u32x4 g0;
      g0[0] = 1u;                                   // count=1, user descriptor
      g0[1] = ksh_lds;                              // lds_addr
      g0[2] = (unsigned)(ga & 0xFFFFFFFFu);         // global_addr[31:0]
      g0[3] = (unsigned)((ga >> 32) & 0x1FFFFFFu)   // global_addr[56:32]
              | (2u << 30);                         // type = 2 ("image")
      i32x8 g1;
      g1[0] = (int)(1u << 16);        // workgroup_mask=0, data_size=1 (2B)
      g1[1] = (int)(8192u << 16);     // tensor_dim0[15:0]=8192 (atomic_bar_addr=0)
      g1[2] = (int)(1u << 16);        // tensor_dim0[31:16]=0, tensor_dim1[15:0]=1
      g1[3] = (int)(8192u << 16);     // tensor_dim1[31:16]=0, tile_dim0=8192
      g1[4] = 1;                      // tile_dim1=1, tile_dim2=0
      g1[5] = 8192;                   // tensor_dim0_stride[31:0]
      g1[6] = 0;                      // stride hi, tensor_dim1_stride lo
      g1[7] = 0;
      i32x4 z4 = {0, 0, 0, 0};
      i32x8 z8 = {0, 0, 0, 0, 0, 0, 0, 0};
      __builtin_amdgcn_tensor_load_to_lds(g0, g1, z4, z4, z8, 0);
      __builtin_amdgcn_s_wait_tensorcnt(0);
    }
#else
    // ---- fallback: manual K chunk staging ----
    for (int u = tid; u < 1024; u += 128) {
      int h0 = u * 8; int key = h0 >> 8; int d = h0 & 255;
      *(uint4*)&ksh[key * 256 + d] =
          *(const uint4*)(kbase + (size_t)(kc + key) * 256 + d);
    }
#endif
    // stage V chunk transposed [d][key]
    for (int u = tid; u < 2048; u += 128) {
      int h0 = u * 4; int key = h0 >> 8; int d = h0 & 255;
      const __bf16* pv = vbase + (size_t)(kc + key) * 256 + d;
#pragma unroll
      for (int j = 0; j < 4; ++j) vsh[(d + j) * 32 + key] = pv[j];
    }
    if (kc + 32 < S) __builtin_prefetch(kbase + (size_t)(kc + 32) * 256, 0, 1);
    __syncthreads();

    // scores: q(16x256) . K_chunk^T -> two 16x16 tiles
    v8f s0 = {}, s1 = {};
#pragma unroll
    for (int kk = 0; kk < 8; ++kk) {
      V16 k0, k1;
      const __bf16* pk0 = &ksh[r16 * 256 + kk * 32 + hi * 16];
      k0.h[0] = *(const v8bf*)pk0; k0.h[1] = *(const v8bf*)(pk0 + 8);
      const __bf16* pk1 = &ksh[(16 + r16) * 256 + kk * 32 + hi * 16];
      k1.h[0] = *(const v8bf*)pk1; k1.h[1] = *(const v8bf*)(pk1 + 8);
      s0 = WMMA_BF16(qA[kk].v, k0.v, s0);
      s1 = WMMA_BF16(qA[kk].v, k1.v, s1);
    }

    // online softmax (C-layout: lane=col, vgpr i = row i + 8*hi)
    float p0[8], p1[8], corr[8];
#pragma unroll
    for (int i = 0; i < 8; ++i) {
      float a = s0[i] * scale, c = s1[i] * scale;
      float mt = fmaxf(a, c);
#pragma unroll
      for (int m = 8; m >= 1; m >>= 1) mt = fmaxf(mt, __shfl_xor(mt, m, 16));
      float mn = fmaxf(mrun[i], mt);
      corr[i] = __expf(mrun[i] - mn);
      p0[i] = __expf(a - mn);
      p1[i] = __expf(c - mn);
      float rs = p0[i] + p1[i];
#pragma unroll
      for (int m = 8; m >= 1; m >>= 1) rs += __shfl_xor(rs, m, 16);
      lrun[i] = lrun[i] * corr[i] + rs;
      mrun[i] = mn;
    }
#pragma unroll
    for (int f = 0; f < 16; ++f)
#pragma unroll
      for (int i = 0; i < 8; ++i) accY[f][i] *= corr[i];

    // P: C-layout -> A-layout via per-wave LDS staging
    __bf16* pw = &psh[wave][0];
#pragma unroll
    for (int i = 0; i < 8; ++i) {
      int r = i + hi * 8;
      pw[r * 32 + r16]      = (__bf16)p0[i];
      pw[r * 32 + 16 + r16] = (__bf16)p1[i];
    }
    asm volatile("s_wait_dscnt 0" ::: "memory");  // same-wave LDS RAW
    V16 pA;
    {
      const __bf16* pp = &pw[r16 * 32 + hi * 16];
      pA.h[0] = *(const v8bf*)pp;
      pA.h[1] = *(const v8bf*)(pp + 8);
    }

    // y(16x256) += P(16x32) . V_chunk(32x256)
#pragma unroll
    for (int f = 0; f < 16; ++f) {
      V16 vb;
      const __bf16* pv = &vsh[(f * 16 + r16) * 32 + hi * 16];
      vb.h[0] = *(const v8bf*)pv;
      vb.h[1] = *(const v8bf*)(pv + 8);
      accY[f] = WMMA_BF16(pA.v, vb.v, accY[f]);
    }
    __syncthreads();
  }

  // epilogue: 1/l, sigmoid gate, store bf16 y in [t][h*256+d]
#pragma unroll
  for (int f = 0; f < 16; ++f) {
#pragma unroll
    for (int i = 0; i < 8; ++i) {
      int srow = q0 + i + hi * 8;
      int d = f * 16 + r16;
      size_t t = (size_t)b * S + srow;
      float y = accY[f][i] / lrun[i];
      float g = (float)qgb[t * 8192 + h * 512 + 256 + d];
      y *= 1.0f / (1.0f + __expf(-g));
      yb[t * 4096 + h * 256 + d] = (__bf16)y;
    }
  }
}

// ----------------------------------------------------------------- launcher
extern "C" void kernel_launch(void* const* d_in, const int* in_sizes, int n_in,
                              void* d_out, int out_size, void* d_ws, size_t ws_size,
                              hipStream_t stream) {
  const float* x    = (const float*)d_in[0];
  const float* rope = (const float*)d_in[1];
  const float* wq   = (const float*)d_in[2];
  const float* wk   = (const float*)d_in[3];
  const float* wv   = (const float*)d_in[4];
  const float* wo   = (const float*)d_in[5];
  const float* qnw  = (const float*)d_in[6];
  const float* knw  = (const float*)d_in[7];
  (void)in_sizes; (void)n_in; (void)out_size; (void)ws_size;

  const int S = 2048, D = 4096, H = 16, KV = 4;
  const long T = 4096;  // B*S

  char* ws = (char*)d_ws;
  size_t off = 0;
  auto take = [&](size_t bytes) -> char* {
    char* p = ws + off; off += (bytes + 255) & ~(size_t)255; return p;
  };
  __bf16* xb  = (__bf16*)take((size_t)T * D * 2);
  __bf16* wqb = (__bf16*)take((size_t)8192 * D * 2);
  __bf16* wkb = (__bf16*)take((size_t)1024 * D * 2);
  __bf16* wvb = (__bf16*)take((size_t)1024 * D * 2);
  __bf16* wob = (__bf16*)take((size_t)D * D * 2);
  __bf16* qgb = (__bf16*)take((size_t)T * 8192 * 2);
  __bf16* krb = (__bf16*)take((size_t)T * 1024 * 2);
  __bf16* vrb = (__bf16*)take((size_t)T * 1024 * 2);
  __bf16* qpb = (__bf16*)take((size_t)T * 4096 * 2);
  __bf16* kpb = (__bf16*)take((size_t)T * 1024 * 2);
  __bf16* vpb = (__bf16*)take((size_t)T * 1024 * 2);
  __bf16* yb  = (__bf16*)take((size_t)T * 4096 * 2);

  // 1) casts to bf16
  cvt_f32_bf16<<<dim3(2048), dim3(256), 0, stream>>>(x,  xb,  T * D);
  cvt_f32_bf16<<<dim3(2048), dim3(256), 0, stream>>>(wq, wqb, (long)8192 * D);
  cvt_f32_bf16<<<dim3(1024), dim3(256), 0, stream>>>(wk, wkb, (long)1024 * D);
  cvt_f32_bf16<<<dim3(1024), dim3(256), 0, stream>>>(wv, wvb, (long)1024 * D);
  cvt_f32_bf16<<<dim3(2048), dim3(256), 0, stream>>>(wo, wob, (long)D * D);

  // 2) projections (NT WMMA GEMMs)
  gemm_nt_bf16<<<dim3(8192 / 128, 4096 / 64), dim3(256), 0, stream>>>(
      xb, wqb, nullptr, qgb, 4096, 8192, D);
  gemm_nt_bf16<<<dim3(1024 / 128, 4096 / 64), dim3(256), 0, stream>>>(
      xb, wkb, nullptr, krb, 4096, 1024, D);
  gemm_nt_bf16<<<dim3(1024 / 128, 4096 / 64), dim3(256), 0, stream>>>(
      xb, wvb, nullptr, vrb, 4096, 1024, D);

  // 3) RMSNorm + RoPE + relayout  (waves = T*heads; 8 waves/block)
  normrope_kernel<<<dim3((unsigned)(T * H / 8)), dim3(256), 0, stream>>>(
      qgb, qnw, rope, qpb, H, 8192, 512, 1, S);
  normrope_kernel<<<dim3((unsigned)(T * KV / 8)), dim3(256), 0, stream>>>(
      krb, knw, rope, kpb, KV, 1024, 256, 1, S);
  normrope_kernel<<<dim3((unsigned)(T * KV / 8)), dim3(256), 0, stream>>>(
      vrb, qnw, rope, vpb, KV, 1024, 256, 0, S);

  // 4) flash attention + gate
  attn_kernel<<<dim3(32, S / 64), dim3(128), 0, stream>>>(
      qpb, kpb, vpb, qgb, yb, S);

  // 5) output projection -> fp32 d_out
  gemm_nt_bf16<<<dim3(4096 / 128, 4096 / 64), dim3(256), 0, stream>>>(
      yb, wob, (float*)d_out, nullptr, 4096, 4096, D);
}
